// PHGT_77369540870680
// MI455X (gfx1250) — matrix operand order repaired and checked by reading, not compiled, not run on hardware
//
#include <hip/hip_runtime.h>
#include <hip/hip_bf16.h>

typedef __attribute__((ext_vector_type(16))) _Float16 v16h_t;
typedef __attribute__((ext_vector_type(8)))  _Float16 v8h_t;
typedef __attribute__((ext_vector_type(8)))  float    v8f_t;

// Problem constants (fixed by the reference)
static constexpr int    kN   = 8192;
static constexpr int    kE   = 262144;
static constexpr int    kIN  = 128;
static constexpr int    kD   = 256;
static constexpr int    kH   = 4;
static constexpr int    kHD  = 64;
static constexpr int    kP   = 32;
static constexpr int    kL   = 4;
static constexpr int    kC   = 8;
static constexpr int    kNC  = 16;
static constexpr int    kS   = 1 + kP + kC;          // 41 tokens
static constexpr long   kSN  = (long)kS * kN;        // 335872 rows
static constexpr float  kScale = 0.125f;             // 1/sqrt(64)

// ---------------------------------------------------------------------------
// WMMA GEMM:  C[M,Ncols] = A[M,K] * B (+bias, opt ReLU)
//   A: f32 row-major, lda=K.  B: f32 [K,Ncols] if !TRANSB, [Ncols,K] if TRANSB.
//   Block: 256 threads (8 waves), tile 128x128, K-step 32.
//   Wave w owns a 32x64 quadrant: 2x4 16x16 tiles -> 8 WMMAs per K-step.
//   Staging: per-thread fixed chunks, float4 global loads -> f16 -> b128/b16 LDS stores.
// ---------------------------------------------------------------------------
#define BM 128
#define BN 128
#define BK 32
#define LDSROW 40   // padded row (80B): keeps 16B alignment, spreads banks

template <int TRANSB>
__global__ __launch_bounds__(256)
void phgt_gemm_wmma(const float* __restrict__ A, const float* __restrict__ B,
                    const float* __restrict__ bias, float* __restrict__ C,
                    int M, int Ncols, int K, int relu) {
  __shared__ __align__(16) _Float16 As[BM][LDSROW];
  __shared__ __align__(16) _Float16 Bs[BN][LDSROW];

  const int lane    = threadIdx.x & 31;
  const int wave    = threadIdx.x >> 5;
  const int wm      = (wave >> 1) * 32;      // 0,32,64,96
  const int wn      = (wave & 1)  * 64;      // 0,64
  const int l15     = lane & 15;
  const int halfSel = lane >> 4;             // 0: lanes 0-15, 1: lanes 16-31
  const int m0      = blockIdx.y * BM;
  const int n0      = blockIdx.x * BN;

  v8f_t acc[2][4] = {};

  for (int k0 = 0; k0 < K; k0 += BK) {
    // ---- stage A tile (128x32): 2 threads per row, 16 contiguous f32 each ----
    {
      const int r = threadIdx.x >> 1;
      const int h = (threadIdx.x & 1) << 4;            // 0 or 16
      const float* gp = A + (size_t)(m0 + r) * K + (k0 + h);
      const float4 f0 = ((const float4*)gp)[0];
      const float4 f1 = ((const float4*)gp)[1];
      const float4 f2 = ((const float4*)gp)[2];
      const float4 f3 = ((const float4*)gp)[3];
      v8h_t p0, p1;
      p0[0]=(_Float16)f0.x; p0[1]=(_Float16)f0.y; p0[2]=(_Float16)f0.z; p0[3]=(_Float16)f0.w;
      p0[4]=(_Float16)f1.x; p0[5]=(_Float16)f1.y; p0[6]=(_Float16)f1.z; p0[7]=(_Float16)f1.w;
      p1[0]=(_Float16)f2.x; p1[1]=(_Float16)f2.y; p1[2]=(_Float16)f2.z; p1[3]=(_Float16)f2.w;
      p1[4]=(_Float16)f3.x; p1[5]=(_Float16)f3.y; p1[6]=(_Float16)f3.z; p1[7]=(_Float16)f3.w;
      *(v8h_t*)&As[r][h]     = p0;
      *(v8h_t*)&As[r][h + 8] = p1;
    }
    // ---- stage B tile as Bs[n][k] ----
    if (TRANSB) {  // B[Ncols,K]: contiguous along k, same pattern as A
      const int n  = threadIdx.x >> 1;
      const int h  = (threadIdx.x & 1) << 4;
      const int gn = n0 + n;
      v8h_t p0 = {}, p1 = {};
      if (gn < Ncols) {
        const float* gp = B + (size_t)gn * K + (k0 + h);
        const float4 f0 = ((const float4*)gp)[0];
        const float4 f1 = ((const float4*)gp)[1];
        const float4 f2 = ((const float4*)gp)[2];
        const float4 f3 = ((const float4*)gp)[3];
        p0[0]=(_Float16)f0.x; p0[1]=(_Float16)f0.y; p0[2]=(_Float16)f0.z; p0[3]=(_Float16)f0.w;
        p0[4]=(_Float16)f1.x; p0[5]=(_Float16)f1.y; p0[6]=(_Float16)f1.z; p0[7]=(_Float16)f1.w;
        p1[0]=(_Float16)f2.x; p1[1]=(_Float16)f2.y; p1[2]=(_Float16)f2.z; p1[3]=(_Float16)f2.w;
        p1[4]=(_Float16)f3.x; p1[5]=(_Float16)f3.y; p1[6]=(_Float16)f3.z; p1[7]=(_Float16)f3.w;
      }
      *(v8h_t*)&Bs[n][h]     = p0;
      *(v8h_t*)&Bs[n][h + 8] = p1;
    } else {       // B[K,Ncols]: contiguous along n, transpose into LDS
      const int k  = threadIdx.x >> 3;                 // 0..31
      const int nb = (threadIdx.x & 7) << 4;           // 0,16,...,112
      const float* gp = B + (size_t)(k0 + k) * Ncols + (n0 + nb);
#pragma unroll
      for (int q = 0; q < 4; ++q) {
        const int gn = n0 + nb + q * 4;
        float4 f = (gn < Ncols) ? ((const float4*)gp)[q]
                                : make_float4(0.f, 0.f, 0.f, 0.f);
        Bs[nb + q * 4 + 0][k] = (_Float16)f.x;
        Bs[nb + q * 4 + 1][k] = (_Float16)f.y;
        Bs[nb + q * 4 + 2][k] = (_Float16)f.z;
        Bs[nb + q * 4 + 3][k] = (_Float16)f.w;
      }
    }
    if (k0 + BK < K)  // global_prefetch_b8 on next A tile
      __builtin_prefetch(&A[(size_t)(m0 + (threadIdx.x & 127)) * K + (k0 + BK)], 0, 1);
    __syncthreads();

    // ---- fragments per documented wave32 layouts ----
    v16h_t afr[2], bfr[4];
#pragma unroll
    for (int mi = 0; mi < 2; ++mi) {
      const _Float16* pr = &As[wm + mi * 16 + l15][0];
      v8h_t lo = *(const v8h_t*)(pr + 8 * halfSel);        // K 0..7  / 8..15
      v8h_t hi = *(const v8h_t*)(pr + 16 + 8 * halfSel);   // K 16..23 / 24..31
#pragma unroll
      for (int e = 0; e < 8; ++e) { afr[mi][e] = lo[e]; afr[mi][8 + e] = hi[e]; }
    }
#pragma unroll
    for (int ni = 0; ni < 4; ++ni) {
      const _Float16* pr = &Bs[wn + ni * 16 + l15][0];
      v8h_t lo = *(const v8h_t*)(pr + 8 * halfSel);        // K 0..7  / 8..15
      v8h_t hi = *(const v8h_t*)(pr + 16 + 8 * halfSel);   // K 16..23 / 24..31
#pragma unroll
      for (int e = 0; e < 8; ++e) { bfr[ni][e] = lo[e]; bfr[ni][8 + e] = hi[e]; }
    }
#pragma unroll
    for (int mi = 0; mi < 2; ++mi)
#pragma unroll
      for (int ni = 0; ni < 4; ++ni)
        acc[mi][ni] = __builtin_amdgcn_wmma_f32_16x16x32_f16(
            false, afr[mi], false, bfr[ni], (short)0, acc[mi][ni], false, false);
    __syncthreads();
  }

  // ---- epilogue: bias (+ReLU); C/D layout: m = r + 8*halfSel ----
#pragma unroll
  for (int mi = 0; mi < 2; ++mi)
#pragma unroll
    for (int ni = 0; ni < 4; ++ni) {
      int n = n0 + wn + ni * 16 + l15;
      if (n < Ncols) {
        float bv = bias[n];
#pragma unroll
        for (int r = 0; r < 8; ++r) {
          int m = m0 + wm + mi * 16 + 8 * halfSel + r;
          float v = acc[mi][ni][r] + bv;
          if (relu) v = fmaxf(v, 0.0f);
          C[(size_t)m * Ncols + n] = v;
        }
      }
    }
}

// ---------------------------------------------------------------------------
// Helper kernels
// ---------------------------------------------------------------------------
__global__ void phgt_fill(float* __restrict__ p, long n, float v) {
  for (long i = blockIdx.x * (long)blockDim.x + threadIdx.x; i < n;
       i += (long)gridDim.x * blockDim.x) p[i] = v;
}

__global__ void phgt_edge_cnt(const int* __restrict__ ei, float* __restrict__ cnt) {
  int e = blockIdx.x * blockDim.x + threadIdx.x;
  if (e < kE) atomicAdd(&cnt[ei[kE + e]], 1.0f);
}

__global__ void phgt_edge_scat(const int* __restrict__ ei, const float* __restrict__ x,
                               float* __restrict__ agg) {
  long total = (long)kE * kIN;
  for (long i = blockIdx.x * (long)blockDim.x + threadIdx.x; i < total;
       i += (long)gridDim.x * blockDim.x) {
    int e = (int)(i >> 7), f = (int)(i & 127);
    atomicAdd(&agg[(size_t)ei[kE + e] * kIN + f], x[(size_t)ei[e] * kIN + f]);
  }
}

// Acat[n, 0:128] = agg/cnt, Acat[n, 128:256] = x
__global__ void phgt_acat(const float* __restrict__ agg, const float* __restrict__ cnt,
                          const float* __restrict__ x, float* __restrict__ Acat) {
  long total = (long)kN * 256;
  for (long i = blockIdx.x * (long)blockDim.x + threadIdx.x; i < total;
       i += (long)gridDim.x * blockDim.x) {
    int n = (int)(i >> 8), k = (int)(i & 255);
    Acat[i] = (k < kIN) ? agg[(size_t)n * kIN + k] / fmaxf(cnt[n], 1.0f)
                        : x[(size_t)n * kIN + (k - kIN)];
  }
}

// Bcat = vstack(Wl[128,256], Wr[128,256]) -> [256,256]
__global__ void phgt_bcat(const float* __restrict__ Wl, const float* __restrict__ Wr,
                          float* __restrict__ Bcat) {
  int i = blockIdx.x * blockDim.x + threadIdx.x;
  if (i < 256 * 256) {
    int k = i >> 8, n = i & 255;
    Bcat[i] = (k < kIN) ? Wl[(size_t)k * kD + n] : Wr[(size_t)(k - kIN) * kD + n];
  }
}

__global__ __launch_bounds__(256)
void phgt_sem(const float* __restrict__ ne, const int* __restrict__ mp,
              float* __restrict__ sem) {
  int p = blockIdx.x, d = threadIdx.x;
  float s = 0.0f;
  for (int l = 0; l < kL; ++l) s += ne[(size_t)mp[p * kL + l] * kD + d];
  sem[(size_t)p * kD + d] = s * 0.25f;
}

__global__ __launch_bounds__(256)
void phgt_glob(const float* __restrict__ ne, const int* __restrict__ cid,
               float* __restrict__ glob) {
  int c = blockIdx.x, d = threadIdx.x;
  float s = 0.0f, cnt = 0.0f;
  for (int n = 0; n < kN; ++n)
    if (cid[n] == c) { s += ne[(size_t)n * kD + d]; cnt += 1.0f; }
  glob[(size_t)c * kD + d] = s / fmaxf(cnt, 1.0f);
}

__global__ void phgt_tokens(const float* __restrict__ ne, const float* __restrict__ sem,
                            const float* __restrict__ glob, float* __restrict__ T) {
  long total = (long)kS * kN * kD;               // 41 * 2^21
  for (long i = blockIdx.x * (long)blockDim.x + threadIdx.x; i < total;
       i += (long)gridDim.x * blockDim.x) {
    int  s   = (int)(i >> 21);
    long rem = i & ((1L << 21) - 1);
    int  d   = (int)(rem & 255);
    float v;
    if (s == 0)            v = ne[rem];
    else if (s < 1 + kP)   v = sem[(size_t)(s - 1) * kD + d];
    else                   v = glob[(size_t)(s - 1 - kP) * kD + d];
    T[i] = v;
  }
}

// Per-(node, head) attention over S=41 tokens, HD=64. VALU (3% of FLOPs).
__global__ __launch_bounds__(64)
void phgt_attn(const float* __restrict__ qkv, float* __restrict__ out) {
  int b = blockIdx.x, h = blockIdx.y, tid = threadIdx.x;
  __shared__ float q[kS][kHD], kk[kS][kHD], vv[kS][kHD], sc[kS][kS];
  for (int i = tid; i < kS * kHD; i += 64) {
    int s = i >> 6, d = i & 63;
    size_t base = ((size_t)s * kN + b) * (3 * kD) + h * kHD + d;
    q[s][d]  = qkv[base];
    kk[s][d] = qkv[base + kD];
    vv[s][d] = qkv[base + 2 * kD];
  }
  __syncthreads();
  for (int i = tid; i < kS * kS; i += 64) {
    int s = i / kS, t = i % kS;
    float a = 0.0f;
    for (int d = 0; d < kHD; ++d) a += q[s][d] * kk[t][d];
    sc[s][t] = a * kScale;
  }
  __syncthreads();
  if (tid < kS) {
    float mx = -1e30f;
    for (int t = 0; t < kS; ++t) mx = fmaxf(mx, sc[tid][t]);
    float sum = 0.0f;
    for (int t = 0; t < kS; ++t) { float e = __expf(sc[tid][t] - mx); sc[tid][t] = e; sum += e; }
    float inv = 1.0f / sum;
    for (int t = 0; t < kS; ++t) sc[tid][t] *= inv;
  }
  __syncthreads();
  for (int i = tid; i < kS * kHD; i += 64) {
    int s = i >> 6, d = i & 63;
    float a = 0.0f;
    for (int t = 0; t < kS; ++t) a += sc[s][t] * vv[t][d];
    out[((size_t)s * kN + b) * kD + h * kHD + d] = a;
  }
}

// out = LayerNorm(a + b) * g + beta   (D=256, one wave32 per row, 8 rows/block)
__global__ __launch_bounds__(256)
void phgt_lnadd(const float* __restrict__ a, const float* __restrict__ b,
                const float* __restrict__ g, const float* __restrict__ beta,
                float* __restrict__ out, long rows) {
  int lane = threadIdx.x & 31;
  long row = (long)blockIdx.x * 8 + (threadIdx.x >> 5);
  if (row >= rows) return;
  const float* pa = a + row * kD;
  const float* pb = b + row * kD;
  float x[8]; float s = 0.0f;
#pragma unroll
  for (int j = 0; j < 8; ++j) { int e = lane + 32 * j; x[j] = pa[e] + pb[e]; s += x[j]; }
  for (int off = 16; off > 0; off >>= 1) s += __shfl_xor(s, off, 32);
  float mean = s * (1.0f / kD);
  float v = 0.0f;
#pragma unroll
  for (int j = 0; j < 8; ++j) { float d = x[j] - mean; v += d * d; }
  for (int off = 16; off > 0; off >>= 1) v += __shfl_xor(v, off, 32);
  float r = rsqrtf(v * (1.0f / kD) + 1e-5f);
#pragma unroll
  for (int j = 0; j < 8; ++j) {
    int e = lane + 32 * j;
    out[row * kD + e] = (x[j] - mean) * r * g[e] + beta[e];
  }
}

// ---------------------------------------------------------------------------
// Host-side orchestration
// ---------------------------------------------------------------------------
static inline void launch_gemm(const float* A, const float* B, const float* bias,
                               float* C, long M, int Ncols, int K, int transB,
                               int relu, hipStream_t s) {
  dim3 grid((Ncols + BN - 1) / BN, (unsigned)(M / BM));
  if (transB)
    phgt_gemm_wmma<1><<<grid, 256, 0, s>>>(A, B, bias, C, (int)M, Ncols, K, relu);
  else
    phgt_gemm_wmma<0><<<grid, 256, 0, s>>>(A, B, bias, C, (int)M, Ncols, K, relu);
}

static void run_layer(float* T, float* X, float* BIG, float* AO, float* PF,
                      const float* const* Lp, hipStream_t s) {
  const float *Wi = Lp[0], *bi = Lp[1], *Wo = Lp[2], *bo = Lp[3];
  const float *g1 = Lp[4], *b1 = Lp[5], *W1 = Lp[6], *bf1 = Lp[7];
  const float *W2 = Lp[8], *bf2 = Lp[9];
  // QKV = T @ Wi^T + bi   (BIG reused as QKV here, as FFN hidden later)
  launch_gemm(T, Wi, bi, BIG, kSN, 3 * kD, kD, /*transB=*/1, 0, s);
  phgt_attn<<<dim3(kN, kH), 64, 0, s>>>(BIG, AO);
  // proj = AO @ Wo^T + bo
  launch_gemm(AO, Wo, bo, PF, kSN, kD, kD, /*transB=*/1, 0, s);
  // X = LN(T + proj)
  phgt_lnadd<<<(unsigned)(kSN / 8), 256, 0, s>>>(T, PF, g1, b1, X, kSN);
  // H = relu(X @ W1 + bf1)
  launch_gemm(X, W1, bf1, BIG, kSN, 4 * kD, kD, 0, /*relu=*/1, s);
  // F = H @ W2 + bf2
  launch_gemm(BIG, W2, bf2, PF, kSN, kD, 4 * kD, 0, 0, s);
  // T = LN(X + F)   (bug-faithful: g1/b1 again)
  phgt_lnadd<<<(unsigned)(kSN / 8), 256, 0, s>>>(X, PF, g1, b1, T, kSN);
}

extern "C" void kernel_launch(void* const* d_in, const int* in_sizes, int n_in,
                              void* d_out, int out_size, void* d_ws, size_t ws_size,
                              hipStream_t stream) {
  (void)in_sizes; (void)n_in; (void)out_size; (void)ws_size;

  const float* x   = (const float*)d_in[0];
  const int*   ei  = (const int*)  d_in[1];
  const int*   mp  = (const int*)  d_in[2];
  const int*   cid = (const int*)  d_in[3];
  const float* sWl = (const float*)d_in[4];
  const float* sbl = (const float*)d_in[5];
  const float* sWr = (const float*)d_in[6];
  const float* L0[10]; const float* L1[10];
  for (int i = 0; i < 10; ++i) { L0[i] = (const float*)d_in[7 + i];
                                 L1[i] = (const float*)d_in[17 + i]; }
  const float* Wc1 = (const float*)d_in[27];
  const float* bc1 = (const float*)d_in[28];
  const float* Wc2 = (const float*)d_in[29];
  const float* bc2 = (const float*)d_in[30];
  float* outp = (float*)d_out;

  // workspace carve (256B aligned)
  char* w = (char*)d_ws;
  auto carve = [&](size_t bytes) -> float* {
    float* p = (float*)w; w += (bytes + 255) & ~(size_t)255; return p;
  };
  float* agg  = carve((size_t)kN * kIN * 4);        // 4 MB
  float* cnt  = carve((size_t)kN * 4);
  float* Acat = carve((size_t)kN * 256 * 4);        // 8 MB
  float* Bcat = carve((size_t)256 * 256 * 4);
  float* ne   = carve((size_t)kN * kD * 4);         // 8 MB
  float* sem  = carve((size_t)kP * kD * 4);
  float* glob = carve((size_t)kC * kD * 4);
  float* T    = carve((size_t)kSN * kD * 4);        // 344 MB
  float* X    = carve((size_t)kSN * kD * 4);        // 344 MB
  float* AO   = carve((size_t)kSN * kD * 4);        // 344 MB
  float* PF   = carve((size_t)kSN * kD * 4);        // 344 MB
  float* BIG  = carve((size_t)kSN * 4 * kD * 4);    // 1.37 GB (QKV / FFN hidden)
  float* clsH = carve((size_t)kN * kD * 4);         // 8 MB

  // 1) SAGE mean aggregation
  phgt_fill<<<4096, 256, 0, stream>>>(agg, (long)kN * kIN, 0.0f);
  phgt_fill<<<32, 256, 0, stream>>>(cnt, (long)kN, 0.0f);
  phgt_edge_cnt<<<(kE + 255) / 256, 256, 0, stream>>>(ei, cnt);
  phgt_edge_scat<<<32768, 256, 0, stream>>>(ei, x, agg);

  // 2) node_emb = [agg/cnt | x] @ [Wl ; Wr] + bl   (fused K=256 WMMA GEMM)
  phgt_acat<<<8192, 256, 0, stream>>>(agg, cnt, x, Acat);
  phgt_bcat<<<(256 * 256 + 255) / 256, 256, 0, stream>>>(sWl, sWr, Bcat);
  launch_gemm(Acat, Bcat, sbl, ne, kN, kD, 256, 0, 0, stream);

  // 3) sem / glob pooling, token assembly
  phgt_sem<<<kP, 256, 0, stream>>>(ne, mp, sem);
  phgt_glob<<<kC, 256, 0, stream>>>(ne, cid, glob);
  phgt_tokens<<<65536, 256, 0, stream>>>(ne, sem, glob, T);

  // 4) transformer layers (each ends with tokens back in T)
  run_layer(T, X, BIG, AO, PF, L0, stream);
  run_layer(T, X, BIG, AO, PF, L1, stream);

  // 5) classifier on node token (first kN*kD block of T)
  launch_gemm(T, Wc1, bc1, clsH, kN, kD, kD, 0, /*relu=*/1, stream);
  launch_gemm(clsH, Wc2, bc2, outp, kN, kNC, kD, 0, 0, stream);
}
